// TransformerQA_64768106823811
// MI455X (gfx1250) — matrix operand order repaired and checked by reading, not compiled, not run on hardware
//
#include <hip/hip_runtime.h>
#include <hip/hip_bf16.h>
#include <cmath>

// ---- model constants (match reference) ----
#define D_MODEL 512
#define NHEAD   8
#define DHEAD   64
#define NLAYER  6
#define DFF     2048
#define NEG_INF (-1e9f)

typedef __attribute__((ext_vector_type(16))) _Float16 v16h;
typedef __attribute__((ext_vector_type(8)))  float    v8f;

// ---------------------------------------------------------------------------
// WMMA fragment loaders (f32 -> f16 fragments), wave32, contiguous k order (not the ISA table order; A and B match).
//  A (16x32, MxK): lane = 16*half + m. half=0 holds K 0..15,
//                  half=1 holds K 16..31; VGPR v pairs K.
//  B (32x16, KxN): lane = 16*half + n. half=0 holds K 0..15, half=1 K 16..31.
// ---------------------------------------------------------------------------
__device__ __forceinline__ void load_a_frag(const float* __restrict__ p, int ld,
                                            int row, int kb, int half, v16h& f) {
  const float* q = p + (size_t)row * ld + kb + 16 * half;
#pragma unroll
  for (int v = 0; v < 8; ++v) {
    int ko = 2 * v;
    f[2 * v]     = (_Float16)q[ko];
    f[2 * v + 1] = (_Float16)q[ko + 1];
  }
}

// A fragment from an LDS-staged f32 tile with padded row stride (floats)
__device__ __forceinline__ void load_a_frag_lds(const float* sh, int ldp,
                                                int r, int half, v16h& f) {
  const float* q = sh + r * ldp + 16 * half;
#pragma unroll
  for (int v = 0; v < 8; ++v) {
    int ko = 2 * v;
    f[2 * v]     = (_Float16)q[ko];
    f[2 * v + 1] = (_Float16)q[ko + 1];
  }
}

// B from row-major [K, N] matrix, column = col
__device__ __forceinline__ void load_b_frag(const float* __restrict__ p, int ld,
                                            int kb, int col, int half, v16h& f) {
  const float* q = p + (size_t)(kb + 16 * half) * ld + col;
#pragma unroll
  for (int v = 0; v < 8; ++v) {
    f[2 * v]     = (_Float16)q[(size_t)(2 * v) * ld];
    f[2 * v + 1] = (_Float16)q[(size_t)(2 * v + 1) * ld];
  }
}

// B = M^T where M is row-major [N, K] (used for Q @ K^T): contiguous 16 floats
__device__ __forceinline__ void load_bt_frag(const float* __restrict__ p, int ld,
                                             int row, int kb, int half, v16h& f) {
  const float* q = p + (size_t)row * ld + kb + 16 * half;
#pragma unroll
  for (int v = 0; v < 8; ++v) {
    f[2 * v]     = (_Float16)q[2 * v];
    f[2 * v + 1] = (_Float16)q[2 * v + 1];
  }
}

// ---------------------------------------------------------------------------
// Generic GEMM: C[M,N] = act(A[M,K] @ B[K,N] + bias), f32 row-major.
// 256 threads = 8 waves; each wave -> 16x32 C tile (two WMMA tiles sharing
// one A fragment); block tile = 16 x 256. Grid: (N/256, M/16).
// The 16x32 A tile (shared by all 8 waves) is staged into a double-buffered
// LDS slot by wave 0 with GLOBAL_LOAD_ASYNC_TO_LDS_B128 (ASYNCcnt), so the
// other waves never touch global memory for A.
// LDS rows padded to 36 floats (144B, 16B aligned) to avoid bank conflicts.
// ---------------------------------------------------------------------------
#define A_LDP 36

__global__ void wmma_gemm_kernel(const float* __restrict__ A, const float* __restrict__ B,
                                 const float* __restrict__ bias, float* __restrict__ C,
                                 int M, int N, int K, int relu) {
  __shared__ float shA[2][16 * A_LDP];

  const int lane = threadIdx.x & 31;
  const int wave = threadIdx.x >> 5;
  const int half = lane >> 4;
  const int r    = lane & 15;
  const int colBase = blockIdx.x * 256 + wave * 32;
  const int rowBase = blockIdx.y * 16;
  const int col0 = colBase + r;
  const int col1 = col0 + 16;

  // lane -> (row, 16B chunk) mapping for the async A-tile copy (wave 0 only)
  const int cp_row   = lane >> 3;        // 0..3 (base row per issue)
  const int cp_chunk = lane & 7;         // 8 x 16B chunks per 128B row
  const unsigned lds0 = (unsigned)(uintptr_t)&shA[0][0];
  const unsigned lds1 = (unsigned)(uintptr_t)&shA[1][0];

  const int nsteps = K >> 5;

  // prologue: wave 0 stages A tile for kb=0 into buffer 0
  if (wave == 0) {
#pragma unroll
    for (int j = 0; j < 4; ++j) {
      int row = j * 4 + cp_row;
      unsigned long long g =
          (unsigned long long)(uintptr_t)(A + (size_t)(rowBase + row) * K + cp_chunk * 4);
      unsigned l = lds0 + (unsigned)(row * A_LDP + cp_chunk * 4) * 4u;
      asm volatile("global_load_async_to_lds_b128 %0, %1, off" :: "v"(l), "v"(g) : "memory");
    }
  }

  v8f acc0 = {}, acc1 = {};
  for (int i = 0; i < nsteps; ++i) {
    const int kb = i << 5;
    if (wave == 0) {
      if (i + 1 < nsteps) {  // stage next tile into the other buffer
        unsigned ldsn = ((i + 1) & 1) ? lds1 : lds0;
#pragma unroll
        for (int j = 0; j < 4; ++j) {
          int row = j * 4 + cp_row;
          unsigned long long g = (unsigned long long)(uintptr_t)(
              A + (size_t)(rowBase + row) * K + (kb + 32) + cp_chunk * 4);
          unsigned l = ldsn + (unsigned)(row * A_LDP + cp_chunk * 4) * 4u;
          asm volatile("global_load_async_to_lds_b128 %0, %1, off" :: "v"(l), "v"(g) : "memory");
        }
        asm volatile("s_wait_asynccnt 0x4" ::: "memory");  // current buffer landed
      } else {
        asm volatile("s_wait_asynccnt 0x0" ::: "memory");
      }
    }
    __syncthreads();  // A tile visible to all waves

    if (i + 1 < nsteps)  // prefetch next B tile rows (global_prefetch_b8)
      __builtin_prefetch(B + (size_t)(kb + 32 + 16 * half) * N + col0, 0, 1);

    const float* shcur = shA[i & 1];
    v16h af, bf0, bf1;
    load_a_frag_lds(shcur, A_LDP, r, half, af);
    load_b_frag(B, N, kb, col0, half, bf0);
    load_b_frag(B, N, kb, col1, half, bf1);
    acc0 = __builtin_amdgcn_wmma_f32_16x16x32_f16(false, af, false, bf0,
                                                  (short)0, acc0, false, false);
    acc1 = __builtin_amdgcn_wmma_f32_16x16x32_f16(false, af, false, bf1,
                                                  (short)0, acc1, false, false);
    __syncthreads();  // tile consumed before wave 0 overwrites this buffer
  }

  const float bv0 = bias ? bias[col0] : 0.0f;
  const float bv1 = bias ? bias[col1] : 0.0f;
#pragma unroll
  for (int v = 0; v < 8; ++v) {
    const size_t orow = (size_t)(rowBase + v + 8 * half) * N;
    float x0 = acc0[v] + bv0;
    float x1 = acc1[v] + bv1;
    if (relu) { x0 = fmaxf(x0, 0.0f); x1 = fmaxf(x1, 0.0f); }
    C[orow + col0] = x0;
    C[orow + col1] = x1;
  }
}

// ---------------------------------------------------------------------------
// Attention scores: S[z, q, k] = (Q_h[q,:] . K_h[k,:]) / 8, masked with -1e9.
// z = b*H + h. Q: [B,Sq,D], K: [B,Sk,D]. Grid: (Sk/128, Sq/16, B*H), block 256.
// ---------------------------------------------------------------------------
__global__ void attn_scores_kernel(const float* __restrict__ Q, const float* __restrict__ Kmat,
                                   float* __restrict__ S,
                                   const int* __restrict__ qtok, const int* __restrict__ ktok,
                                   int Sq, int Sk, int causal) {
  const int lane = threadIdx.x & 31;
  const int wave = threadIdx.x >> 5;
  const int half = lane >> 4;
  const int r    = lane & 15;
  const int z = blockIdx.z;
  const int b = z / NHEAD;
  const int h = z % NHEAD;
  const int colBase = blockIdx.x * 128 + wave * 16;  // key position
  const int rowBase = blockIdx.y * 16;               // query position
  const float* Qh = Q    + (size_t)b * Sq * D_MODEL + h * DHEAD;
  const float* Kh = Kmat + (size_t)b * Sk * D_MODEL + h * DHEAD;
  v8f acc = {};
#pragma unroll
  for (int kb = 0; kb < DHEAD; kb += 32) {
    v16h af, bf;
    load_a_frag(Qh, D_MODEL, rowBase + r, kb, half, af);
    load_bt_frag(Kh, D_MODEL, colBase + r, kb, half, bf);  // B = K^T
    acc = __builtin_amdgcn_wmma_f32_16x16x32_f16(false, af, false, bf,
                                                 (short)0, acc, false, false);
  }
  const int kpos = colBase + r;
  const bool kmask = ktok && (ktok[b * Sk + kpos] == 0);
#pragma unroll
  for (int v = 0; v < 8; ++v) {
    int q = rowBase + v + 8 * half;
    float val = acc[v] * 0.125f;  // 1/sqrt(64)
    bool masked = kmask;
    if (qtok && qtok[b * Sq + q] == 0) masked = true;
    if (causal && kpos > q) masked = true;
    S[((size_t)z * Sq + q) * Sk + kpos] = masked ? NEG_INF : val;
  }
}

// ---------------------------------------------------------------------------
// Row softmax, one wave32 per row. Grid: rows/8, block 256.
// ---------------------------------------------------------------------------
__global__ void softmax_kernel(float* __restrict__ S, int rows, int cols) {
  const int lane = threadIdx.x & 31;
  const int row  = blockIdx.x * 8 + (threadIdx.x >> 5);
  if (row >= rows) return;
  float* p = S + (size_t)row * cols;
  float mx = -INFINITY;
  for (int i = lane; i < cols; i += 32) mx = fmaxf(mx, p[i]);
#pragma unroll
  for (int o = 16; o; o >>= 1) mx = fmaxf(mx, __shfl_xor(mx, o, 32));
  float sum = 0.0f;
  for (int i = lane; i < cols; i += 32) { float e = __expf(p[i] - mx); p[i] = e; sum += e; }
#pragma unroll
  for (int o = 16; o; o >>= 1) sum += __shfl_xor(sum, o, 32);
  const float inv = 1.0f / sum;
  for (int i = lane; i < cols; i += 32) p[i] *= inv;
}

// ---------------------------------------------------------------------------
// O_h = P @ V_h per (b,h); writes head slice into [B,Sq,D] (free concat).
// 128 threads = 4 waves covering DHEAD=64 columns. Grid: (1, Sq/16, B*H).
// ---------------------------------------------------------------------------
__global__ void attn_av_kernel(const float* __restrict__ P, const float* __restrict__ V,
                               float* __restrict__ O, int Sq, int Sk) {
  const int lane = threadIdx.x & 31;
  const int wave = threadIdx.x >> 5;  // 0..3
  const int half = lane >> 4;
  const int r    = lane & 15;
  const int z = blockIdx.z;
  const int b = z / NHEAD;
  const int h = z % NHEAD;
  const int colBase = wave * 16;
  const int rowBase = blockIdx.y * 16;
  const float* Ph = P + (size_t)z * Sq * Sk;
  const float* Vh = V + (size_t)b * Sk * D_MODEL + h * DHEAD;
  v8f acc = {};
  for (int kb = 0; kb < Sk; kb += 32) {
    v16h af, bf;
    load_a_frag(Ph, Sk, rowBase + r, kb, half, af);
    load_b_frag(Vh, D_MODEL, kb, colBase + r, half, bf);
    acc = __builtin_amdgcn_wmma_f32_16x16x32_f16(false, af, false, bf,
                                                 (short)0, acc, false, false);
  }
#pragma unroll
  for (int v = 0; v < 8; ++v) {
    int q = rowBase + v + 8 * half;
    O[((size_t)b * Sq + q) * D_MODEL + h * DHEAD + colBase + r] = acc[v];
  }
}

// ---------------------------------------------------------------------------
// out = LayerNorm(x + a) * g + b, D=512, one wave32 per row. Grid: rows/8.
// ---------------------------------------------------------------------------
__global__ void add_ln_kernel(const float* __restrict__ X, const float* __restrict__ A,
                              const float* __restrict__ g, const float* __restrict__ bb,
                              float* __restrict__ Out, int rows) {
  const int lane = threadIdx.x & 31;
  const int row  = blockIdx.x * 8 + (threadIdx.x >> 5);
  if (row >= rows) return;
  const float* xp = X + (size_t)row * D_MODEL;
  const float* ap = A + (size_t)row * D_MODEL;
  float vals[16];
  float s1 = 0.0f, s2 = 0.0f;
#pragma unroll
  for (int i = 0; i < 16; ++i) {
    int d = lane + i * 32;
    float v = xp[d] + ap[d];
    vals[i] = v; s1 += v; s2 += v * v;
  }
#pragma unroll
  for (int o = 16; o; o >>= 1) { s1 += __shfl_xor(s1, o, 32); s2 += __shfl_xor(s2, o, 32); }
  const float mean = s1 * (1.0f / D_MODEL);
  const float var  = s2 * (1.0f / D_MODEL) - mean * mean;
  const float rs   = rsqrtf(var + 1e-5f);
  float* op = Out + (size_t)row * D_MODEL;
#pragma unroll
  for (int i = 0; i < 16; ++i) {
    int d = lane + i * 32;
    op[d] = (vals[i] - mean) * rs * g[d] + bb[d];
  }
}

// ---------------------------------------------------------------------------
// x[row, :] = emb[tok[row]] * sqrt(D) + pe[row % S, :]. Grid: rows, block 256.
// ---------------------------------------------------------------------------
__global__ void embed_kernel(const int* __restrict__ tok, const float* __restrict__ emb,
                             const float* __restrict__ pe, float* __restrict__ out, int S) {
  const int row = blockIdx.x;
  const int s   = row % S;
  const int t   = tok[row];
  const float scale = 22.62741699796952f;  // sqrt(512)
  for (int d = threadIdx.x; d < D_MODEL; d += blockDim.x)
    out[(size_t)row * D_MODEL + d] =
        emb[(size_t)t * D_MODEL + d] * scale + pe[(size_t)s * D_MODEL + d];
}

// ---------------------------------------------------------------------------
// Host-side orchestration
// ---------------------------------------------------------------------------
static inline void launch_gemm(const float* A, const float* B, const float* bias, float* C,
                               int M, int N, int K, int relu, hipStream_t s) {
  dim3 grid(N / 256, M / 16);
  wmma_gemm_kernel<<<grid, 256, 0, s>>>(A, B, bias, C, M, N, K, relu);
}

extern "C" void kernel_launch(void* const* d_in, const int* in_sizes, int n_in,
                              void* d_out, int out_size, void* d_ws, size_t ws_size,
                              hipStream_t stream) {
  (void)in_sizes; (void)n_in; (void)out_size; (void)ws_size;
#define FP(i) ((const float*)d_in[(i)])
  const int* src = (const int*)d_in[0];
  const int* tgt = (const int*)d_in[1];
  const float* emb = FP(2);
  const float* pe  = FP(3);
  // enc params: 4..19 ; dec params: 20..45 ; fc: 46,47 (insertion order)
  const size_t WSTR  = (size_t)D_MODEL * D_MODEL;
  const size_t BSTR  = D_MODEL;
  const size_t W1STR = (size_t)D_MODEL * DFF;
  const size_t B1STR = DFF;

  const int Bn = 4, Senc = 512, Tdec = 128;
  const int Me = Bn * Senc;  // 2048 encoder rows
  const int Md = Bn * Tdec;  // 512  decoder rows
  const int ZH = Bn * NHEAD; // 32 (b,h) pairs

  // workspace layout (f32)
  float* ws = (float*)d_ws;
  size_t o = 0;
  float* Xe = ws + o; o += (size_t)Me * D_MODEL;   // encoder activations (= memory)
  float* Xd = ws + o; o += (size_t)Md * D_MODEL;   // decoder activations
  float* Qb = ws + o; o += (size_t)Me * D_MODEL;
  float* Kb = ws + o; o += (size_t)Me * D_MODEL;
  float* Vb = ws + o; o += (size_t)Me * D_MODEL;
  float* Ob = ws + o; o += (size_t)Me * D_MODEL;   // concat attn out
  float* Ab = ws + o; o += (size_t)Me * D_MODEL;   // proj / ffn out
  float* Fb = ws + o; o += (size_t)Me * DFF;       // ffn hidden
  float* Sb = ws + o; o += (size_t)ZH * Senc * Senc; // scores (max)

  // ---------------- Encoder ----------------
  embed_kernel<<<Me, 256, 0, stream>>>(src, emb, pe, Xe, Senc);
  for (int l = 0; l < NLAYER; ++l) {
    launch_gemm(Xe, FP(4) + l * WSTR, FP(8)  + l * BSTR, Qb, Me, D_MODEL, D_MODEL, 0, stream);
    launch_gemm(Xe, FP(5) + l * WSTR, FP(9)  + l * BSTR, Kb, Me, D_MODEL, D_MODEL, 0, stream);
    launch_gemm(Xe, FP(6) + l * WSTR, FP(10) + l * BSTR, Vb, Me, D_MODEL, D_MODEL, 0, stream);
    attn_scores_kernel<<<dim3(Senc / 128, Senc / 16, ZH), 256, 0, stream>>>(
        Qb, Kb, Sb, nullptr, src, Senc, Senc, 0);
    softmax_kernel<<<(ZH * Senc) / 8, 256, 0, stream>>>(Sb, ZH * Senc, Senc);
    attn_av_kernel<<<dim3(1, Senc / 16, ZH), 128, 0, stream>>>(Sb, Vb, Ob, Senc, Senc);
    launch_gemm(Ob, FP(7) + l * WSTR, FP(11) + l * BSTR, Ab, Me, D_MODEL, D_MODEL, 0, stream);
    add_ln_kernel<<<Me / 8, 256, 0, stream>>>(Xe, Ab, FP(16) + l * BSTR, FP(17) + l * BSTR, Xe, Me);
    launch_gemm(Xe, FP(12) + l * W1STR, FP(13) + l * B1STR, Fb, Me, DFF, D_MODEL, 1, stream);
    launch_gemm(Fb, FP(14) + l * W1STR, FP(15) + l * BSTR,  Ab, Me, D_MODEL, DFF, 0, stream);
    add_ln_kernel<<<Me / 8, 256, 0, stream>>>(Xe, Ab, FP(18) + l * BSTR, FP(19) + l * BSTR, Xe, Me);
  }

  // ---------------- Decoder ----------------
  embed_kernel<<<Md, 256, 0, stream>>>(tgt, emb, pe, Xd, Tdec);
  for (int l = 0; l < NLAYER; ++l) {
    // self-attention (causal + tgt padding)
    launch_gemm(Xd, FP(20) + l * WSTR, FP(24) + l * BSTR, Qb, Md, D_MODEL, D_MODEL, 0, stream);
    launch_gemm(Xd, FP(21) + l * WSTR, FP(25) + l * BSTR, Kb, Md, D_MODEL, D_MODEL, 0, stream);
    launch_gemm(Xd, FP(22) + l * WSTR, FP(26) + l * BSTR, Vb, Md, D_MODEL, D_MODEL, 0, stream);
    attn_scores_kernel<<<dim3(Tdec / 128, Tdec / 16, ZH), 256, 0, stream>>>(
        Qb, Kb, Sb, tgt, tgt, Tdec, Tdec, 1);
    softmax_kernel<<<(ZH * Tdec) / 8, 256, 0, stream>>>(Sb, ZH * Tdec, Tdec);
    attn_av_kernel<<<dim3(1, Tdec / 16, ZH), 128, 0, stream>>>(Sb, Vb, Ob, Tdec, Tdec);
    launch_gemm(Ob, FP(23) + l * WSTR, FP(27) + l * BSTR, Ab, Md, D_MODEL, D_MODEL, 0, stream);
    add_ln_kernel<<<Md / 8, 256, 0, stream>>>(Xd, Ab, FP(40) + l * BSTR, FP(41) + l * BSTR, Xd, Md);
    // cross-attention (keys/values from encoder memory, src padding mask)
    launch_gemm(Xd, FP(28) + l * WSTR, FP(32) + l * BSTR, Qb, Md, D_MODEL, D_MODEL, 0, stream);
    launch_gemm(Xe, FP(29) + l * WSTR, FP(33) + l * BSTR, Kb, Me, D_MODEL, D_MODEL, 0, stream);
    launch_gemm(Xe, FP(30) + l * WSTR, FP(34) + l * BSTR, Vb, Me, D_MODEL, D_MODEL, 0, stream);
    attn_scores_kernel<<<dim3(Senc / 128, Tdec / 16, ZH), 256, 0, stream>>>(
        Qb, Kb, Sb, nullptr, src, Tdec, Senc, 0);
    softmax_kernel<<<(ZH * Tdec) / 8, 256, 0, stream>>>(Sb, ZH * Tdec, Senc);
    attn_av_kernel<<<dim3(1, Tdec / 16, ZH), 128, 0, stream>>>(Sb, Vb, Ob, Tdec, Senc);
    launch_gemm(Ob, FP(31) + l * WSTR, FP(35) + l * BSTR, Ab, Md, D_MODEL, D_MODEL, 0, stream);
    add_ln_kernel<<<Md / 8, 256, 0, stream>>>(Xd, Ab, FP(42) + l * BSTR, FP(43) + l * BSTR, Xd, Md);
    // FFN
    launch_gemm(Xd, FP(36) + l * W1STR, FP(37) + l * B1STR, Fb, Md, DFF, D_MODEL, 1, stream);
    launch_gemm(Fb, FP(38) + l * W1STR, FP(39) + l * BSTR,  Ab, Md, D_MODEL, DFF, 0, stream);
    add_ln_kernel<<<Md / 8, 256, 0, stream>>>(Xd, Ab, FP(44) + l * BSTR, FP(45) + l * BSTR, Xd, Md);
  }

  // ---------------- Final projection to vocab ----------------
  launch_gemm(Xd, FP(46), FP(47), (float*)d_out, Md, 32000, D_MODEL, 0, stream);
#undef FP
}
